// multiheaded_74371653697775
// MI455X (gfx1250) — matrix operand order repaired and checked by reading, not compiled, not run on hardware
//
#include <hip/hip_runtime.h>

typedef __attribute__((ext_vector_type(16))) __bf16 v16bf;
typedef __attribute__((ext_vector_type(8)))  __bf16 bf16x8;
typedef __attribute__((ext_vector_type(4)))  __bf16 bf16x4;
typedef __attribute__((ext_vector_type(8)))  float  v8f;
typedef __attribute__((ext_vector_type(4)))  int    v4i;

#define DIMX    1024
#define HEADS   8
#define DH      128
#define KP      256
#define NNODES  1024
#define NEDGES  32768

// ---- gfx1250 async load-to-LDS path (ASYNCcnt), with sync fallback so both the host
// pass and any toolchain lacking the builtins still compile.
#if defined(__gfx1250__) && __has_builtin(__builtin_amdgcn_global_load_async_to_lds_b128) && \
    __has_builtin(__builtin_amdgcn_s_wait_asynccnt)
#define USE_ASYNC 1
#else
#define USE_ASYNC 0
#endif

__device__ __forceinline__ void async_cp16(const __bf16* gp, __bf16* lp)
{
#if USE_ASYNC
  __builtin_amdgcn_global_load_async_to_lds_b128(
      (__attribute__((address_space(1))) v4i*)gp,
      (__attribute__((address_space(3))) v4i*)lp,
      0, 0);
#else
  *(bf16x8*)lp = *(const bf16x8*)gp;
#endif
}

__device__ __forceinline__ void wait_async_leq8()
{
#if USE_ASYNC
  __builtin_amdgcn_s_wait_asynccnt(8);
#endif
}

__device__ __forceinline__ void wait_async_zero()
{
#if USE_ASYNC
  __builtin_amdgcn_s_wait_asynccnt(0);
#endif
}

// ---------------- WMMA GEMM: C[M,N] = A[M,K] * B[K,N] (+ bias) ----------------
// A is bf16 [M][K] (lda), B supplied transposed as Bt[N][K] bf16.
// Outputs: Cf (f32, may be null) and/or Cb (bf16, may be null), both with ldc.
// Block: 128 threads = 4 waves (2x2), each wave owns a 64x64 register tile (4x4 frags,
// 16 v_wmma per K-step at 1.0 ds_load_b128 per WMMA).
#define TM  128
#define TN  128
#define TKK 32
#define LDT 40   // padded LDS row stride (bf16): 80B rows keep 16B alignment, dodge conflicts

union V16U { v16bf v; bf16x8 h[2]; };

__global__ __launch_bounds__(128) void gemm_bf16_wmma(
    const __bf16* __restrict__ A, int lda,
    const __bf16* __restrict__ Bt,       // [N][K]
    float* __restrict__ Cf, __bf16* __restrict__ Cb, int ldc,
    const float* __restrict__ bias,      // may be nullptr
    int M, int N, int K)
{
  __shared__ __bf16 As[2][TM * LDT];
  __shared__ __bf16 Bs[2][TN * LDT];

  const int t    = threadIdx.x;
  const int lane = t & 31;
  const int wave = t >> 5;            // 4 waves
  const int wm   = (wave >> 1) * 64;  // 2 waves tile M
  const int wn   = (wave & 1) * 64;   // 2 waves tile N
  const int half = lane >> 4;
  const int l16  = lane & 15;

  const int m0 = blockIdx.y * TM;
  const int n0 = blockIdx.x * TN;

  v8f acc[4][4] = {};

  // per tile: 128 rows x 32 bf16 = 4x16B chunks/row; 512 chunks / 128 thr = 4 each (A and B)
  auto issue_tile = [&](int kk, int buf) {
    #pragma unroll
    for (int r = 0; r < 4; ++r) {
      int id  = t + r * 128;
      int row = id >> 2;
      int c8  = (id & 3) << 3;
      async_cp16(A  + (size_t)(m0 + row) * lda + kk + c8, &As[buf][row * LDT + c8]);
      async_cp16(Bt + (size_t)(n0 + row) * K   + kk + c8, &Bs[buf][row * LDT + c8]);
    }
  };

  auto compute = [&](int cur) {
    const __bf16* Ab = &As[cur][0];
    const __bf16* Bb = &Bs[cur][0];
    V16U a[4], b[4];
    #pragma unroll
    for (int i2 = 0; i2 < 4; ++i2) {  // A frag: K = half*8+0..7 and 16+half*8+0..7
      const __bf16* p = &Ab[(wm + i2 * 16 + l16) * LDT];
      a[i2].h[0] = *(const bf16x8*)(p + half * 8);
      a[i2].h[1] = *(const bf16x8*)(p + 16 + half * 8);
    }
    #pragma unroll
    for (int j = 0; j < 4; ++j) {     // B frag: K = half*8+0..7 and 16+half*8+0..7
      const __bf16* p = &Bb[(wn + j * 16 + l16) * LDT];
      b[j].h[0] = *(const bf16x8*)(p + half * 8);
      b[j].h[1] = *(const bf16x8*)(p + 16 + half * 8);
    }
    #pragma unroll
    for (int i2 = 0; i2 < 4; ++i2)
      #pragma unroll
      for (int j = 0; j < 4; ++j)
        acc[i2][j] = __builtin_amdgcn_wmma_f32_16x16x32_bf16(
            false, a[i2].v, false, b[j].v, (short)0, acc[i2][j], false, false);
  };

  const int nt = K / TKK;
  issue_tile(0, 0);

  int i = 0;
  for (; i + 1 < nt; ++i) {           // steady state: branch-free pipeline body
    issue_tile((i + 1) * TKK, (i + 1) & 1);
    wait_async_leq8();                // only the 8 just-issued copies may be outstanding
    __syncthreads();                  // all waves' LDS writes for buf[i&1] visible
    compute(i & 1);
    __syncthreads();                  // compute done before this buffer is overwritten
  }
  wait_async_zero();
  __syncthreads();
  compute(i & 1);

  // epilogue: lane holds col n=l16; rows m = frag + half*8 + vgpr
  #pragma unroll
  for (int i2 = 0; i2 < 4; ++i2) {
    int mbase = m0 + wm + i2 * 16 + half * 8;
    #pragma unroll
    for (int j = 0; j < 4; ++j) {
      int nn = n0 + wn + j * 16 + l16;
      float bv = bias ? bias[nn] : 0.0f;
      #pragma unroll
      for (int v = 0; v < 8; ++v) {
        float val = acc[i2][j][v] + bv;
        if (Cf) Cf[(size_t)(mbase + v) * ldc + nn] = val;
        if (Cb) Cb[(size_t)(mbase + v) * ldc + nn] = (__bf16)val;
      }
    }
  }
}

// ---------------- transpose + convert to bf16: dst[C][R] = (bf16) src[R][C] ----------------
__global__ __launch_bounds__(256) void transpose_to_bf16(
    const float* __restrict__ src, __bf16* __restrict__ dst, int R, int C)
{
  __shared__ float tile[32][33];
  int bx = blockIdx.x * 32;   // along C
  int by = blockIdx.y * 32;   // along R
  int tx = threadIdx.x;       // 0..31
  int ty = threadIdx.y;       // 0..7
  for (int r = ty; r < 32; r += 8)
    tile[r][tx] = src[(size_t)(by + r) * C + bx + tx];
  __syncthreads();
  for (int r = ty; r < 32; r += 8)
    dst[(size_t)(bx + r) * R + by + tx] = (__bf16)tile[tx][r];
}

// ---------------- elementwise converts ----------------
__global__ void to_bf16(const float* __restrict__ s, __bf16* __restrict__ d, int n)
{
  int i = blockIdx.x * 256 + threadIdx.x;
  if (i < n) d[i] = (__bf16)s[i];
}

__global__ __launch_bounds__(256) void to_bf16_vec(
    const float* __restrict__ s, __bf16* __restrict__ d, int n)  // n multiple of 4
{
  int i = (blockIdx.x * 256 + threadIdx.x) * 4;
  if (i < n) {
    float4 v = *(const float4*)(s + i);
    bf16x4 o = { (__bf16)v.x, (__bf16)v.y, (__bf16)v.z, (__bf16)v.w };
    *(bf16x4*)(d + i) = o;
  }
}

// ---------------- kproj[k][d] = sum_n kv[n][d] * proj_k[n][k]  (tiny, 67 MFLOP) ----------------
__global__ __launch_bounds__(128) void kproj_kernel(
    const float* __restrict__ kv, const float* __restrict__ pk, float* __restrict__ out)
{
  int k = blockIdx.x;   // 0..255
  int d = threadIdx.x;  // 0..127
  float acc = 0.f;
  for (int n = 0; n < NNODES; ++n)
    acc = fmaf(kv[n * DH + d], pk[n * KP + k], acc);
  out[k * DH + d] = acc;
}

// ---------------- softmax rows of 256 (applies 1/sqrt(DH)), writes bf16 attn ----------------
__global__ __launch_bounds__(256) void softmax_rows(
    const float* __restrict__ dots, __bf16* __restrict__ attn)
{
  __shared__ float red[256];
  const int row = blockIdx.x;
  const int t = threadIdx.x;
  const float scale = 0.08838834764831845f;  // 128^-0.5
  float v = dots[(size_t)row * KP + t] * scale;
  red[t] = v; __syncthreads();
  for (int s = 128; s > 0; s >>= 1) { if (t < s) red[t] = fmaxf(red[t], red[t + s]); __syncthreads(); }
  float mx = red[0]; __syncthreads();
  float e = __expf(v - mx);
  red[t] = e; __syncthreads();
  for (int s = 128; s > 0; s >>= 1) { if (t < s) red[t] += red[t + s]; __syncthreads(); }
  float inv = 1.0f / red[0];
  attn[(size_t)row * KP + t] = (__bf16)(e * inv);
}

// ---------------- fused LayerNorm * gamma + beta + residual, in place (rows of 1024) --------
__global__ __launch_bounds__(256) void ln_residual(
    float* __restrict__ x, const float* __restrict__ res,
    const float* __restrict__ gamma, const float* __restrict__ beta)
{
  __shared__ float red[256];
  const int row = blockIdx.x;
  const int t = threadIdx.x;
  float* xr = x + (size_t)row * DIMX;
  float4 v = *(const float4*)(xr + t * 4);
  red[t] = v.x + v.y + v.z + v.w; __syncthreads();
  for (int s = 128; s > 0; s >>= 1) { if (t < s) red[t] += red[t + s]; __syncthreads(); }
  float mu = red[0] * (1.0f / DIMX); __syncthreads();
  float dx = v.x - mu, dy = v.y - mu, dz = v.z - mu, dw = v.w - mu;
  red[t] = dx * dx + dy * dy + dz * dz + dw * dw; __syncthreads();
  for (int s = 128; s > 0; s >>= 1) { if (t < s) red[t] += red[t + s]; __syncthreads(); }
  float rstd = rsqrtf(red[0] * (1.0f / DIMX) + 1e-5f);
  float4 g  = *(const float4*)(gamma + t * 4);
  float4 b  = *(const float4*)(beta + t * 4);
  float4 rr = *(const float4*)(res + (size_t)row * DIMX + t * 4);
  float4 o;
  o.x = dx * rstd * g.x + b.x + rr.x;
  o.y = dy * rstd * g.y + b.y + rr.y;
  o.z = dz * rstd * g.z + b.z + rr.z;
  o.w = dw * rstd * g.w + b.w + rr.w;
  *(float4*)(xr + t * 4) = o;
}

// ======================================================================================
extern "C" void kernel_launch(void* const* d_in, const int* in_sizes, int n_in,
                              void* d_out, int out_size, void* d_ws, size_t ws_size,
                              hipStream_t stream)
{
  const float* node = (const float*)d_in[0];
  const float* edge = (const float*)d_in[1];
  const float* w_q  = (const float*)d_in[2];
  const float* b_q  = (const float*)d_in[3];
  const float* w_e  = (const float*)d_in[4];
  const float* b_e  = (const float*)d_in[5];
  const float* w_tq = (const float*)d_in[6];
  const float* w_tk = (const float*)d_in[7];
  const float* pk   = (const float*)d_in[8];
  const float* w_o  = (const float*)d_in[9];
  const float* b_o  = (const float*)d_in[10];
  const float* g_n  = (const float*)d_in[11];
  const float* be_n = (const float*)d_in[12];
  const float* g_e  = (const float*)d_in[13];
  const float* be_e = (const float*)d_in[14];

  float* x_out = (float*)d_out;                            // [1024,1024]
  float* y_out = (float*)d_out + (size_t)NNODES * DIMX;    // [32768,1024]

  // ---- workspace carve-out
  char* ws = (char*)d_ws;
  size_t off = 0;
  auto carve = [&](size_t bytes) -> void* {
    void* p = ws + off;
    off += (bytes + 255) & ~(size_t)255;
    return p;
  };
  __bf16* WqT   = (__bf16*)carve(sizeof(__bf16) * DIMX * DIMX);   // w_query^T  [1024][1024]
  __bf16* WeT   = (__bf16*)carve(sizeof(__bf16) * DIMX * DIMX);   // w_edge^T
  __bf16* WtqT  = (__bf16*)carve(sizeof(__bf16) * DIMX * DIMX);   // w_to_q^T
  __bf16* WtkT  = (__bf16*)carve(sizeof(__bf16) * DH * DIMX);     // w_to_k^T   [128][1024]
  __bf16* WoT   = (__bf16*)carve(sizeof(__bf16) * DIMX * DIMX);   // w_out^T
  __bf16* nodeB = (__bf16*)carve(sizeof(__bf16) * NNODES * DIMX); // node bf16
  __bf16* edgeB = (__bf16*)carve(sizeof(__bf16) * (size_t)NEDGES * DIMX); // edge bf16 (64MB, L2-resident)
  __bf16* xqB   = (__bf16*)carve(sizeof(__bf16) * NNODES * DIMX);
  __bf16* qB    = (__bf16*)carve(sizeof(__bf16) * NNODES * DIMX);
  float*  kv    = (float*)carve(sizeof(float) * NNODES * DH);
  float*  kpr   = (float*)carve(sizeof(float) * KP * DH);
  __bf16* kprB  = (__bf16*)carve(sizeof(__bf16) * KP * DH);       // [256][128]
  __bf16* kprTB = (__bf16*)carve(sizeof(__bf16) * DH * KP);       // [128][256]
  float*  dots  = (float*)carve(sizeof(float) * HEADS * NNODES * KP);
  __bf16* attnB = (__bf16*)carve(sizeof(__bf16) * HEADS * NNODES * KP);
  __bf16* obufB = (__bf16*)carve(sizeof(__bf16) * NNODES * DIMX);
  (void)ws_size; (void)in_sizes; (void)n_in; (void)out_size;

  dim3 tblk(32, 8);
  dim3 gblk(256);
  dim3 mblk(128);
  auto gemm = [&](const __bf16* A, int lda, const __bf16* Bt,
                  float* Cf, __bf16* Cb, int ldc, const float* bias,
                  int M, int N, int K) {
    dim3 grid(N / TN, M / TM);
    gemm_bf16_wmma<<<grid, mblk, 0, stream>>>(A, lda, Bt, Cf, Cb, ldc, bias, M, N, K);
  };

  // 1) one-time bf16 staging: weights transposed, activations row-major
  transpose_to_bf16<<<dim3(DIMX/32, DIMX/32), tblk, 0, stream>>>(w_q,  WqT,  DIMX, DIMX);
  transpose_to_bf16<<<dim3(DIMX/32, DIMX/32), tblk, 0, stream>>>(w_e,  WeT,  DIMX, DIMX);
  transpose_to_bf16<<<dim3(DIMX/32, DIMX/32), tblk, 0, stream>>>(w_tq, WtqT, DIMX, DIMX);
  transpose_to_bf16<<<dim3(DH/32,   DIMX/32), tblk, 0, stream>>>(w_tk, WtkT, DIMX, DH);
  transpose_to_bf16<<<dim3(DIMX/32, DIMX/32), tblk, 0, stream>>>(w_o,  WoT,  DIMX, DIMX);
  to_bf16_vec<<<dim3((NNODES * DIMX / 4 + 255) / 256), gblk, 0, stream>>>(node, nodeB, NNODES * DIMX);
  to_bf16_vec<<<dim3((int)(((size_t)NEDGES * DIMX / 4 + 255) / 256)), gblk, 0, stream>>>(
      edge, edgeB, NEDGES * DIMX);

  // 2) node path GEMMs (bf16 in, bf16/f32 out as needed)
  gemm(nodeB, DIMX, WqT,  nullptr, xqB, DIMX, b_q,     NNODES, DIMX, DIMX); // xq (bf16 only)
  gemm(xqB,   DIMX, WtqT, nullptr, qB,  DIMX, nullptr, NNODES, DIMX, DIMX); // q  (bf16 only)
  gemm(xqB,   DIMX, WtkT, kv, nullptr,  DH,   nullptr, NNODES, DH,   DIMX); // kv (f32)

  // 3) kproj = proj_k^T @ kv (tiny), then bf16 copies
  kproj_kernel<<<dim3(KP), dim3(DH), 0, stream>>>(kv, pk, kpr);
  to_bf16<<<dim3((KP * DH + 255) / 256), gblk, 0, stream>>>(kpr, kprB, KP * DH);
  transpose_to_bf16<<<dim3(DH/32, KP/32), tblk, 0, stream>>>(kpr, kprTB, KP, DH);

  // 4) attention: dots, softmax(->bf16), out
  for (int h = 0; h < HEADS; ++h)
    gemm(qB + h * DH, DIMX, kprB, dots + (size_t)h * NNODES * KP, nullptr, KP,
         nullptr, NNODES, KP, DH);
  softmax_rows<<<dim3(HEADS * NNODES), gblk, 0, stream>>>(dots, attnB);
  for (int h = 0; h < HEADS; ++h)
    gemm(attnB + (size_t)h * NNODES * KP, KP, kprTB, nullptr, obufB + h * DH, DIMX,
         nullptr, NNODES, DH, KP);

  // 5) output projections straight into d_out
  gemm(obufB, DIMX, WoT, x_out, nullptr, DIMX, b_o, NNODES, DIMX, DIMX);  // x_attn
  gemm(edgeB, DIMX, WeT, y_out, nullptr, DIMX, b_e, NEDGES, DIMX, DIMX);  // edge (dominant)

  // 6) fused LayerNorm + residual, in place
  ln_residual<<<dim3(NNODES), gblk, 0, stream>>>(x_out, node, g_n, be_n);
  ln_residual<<<dim3(NEDGES), gblk, 0, stream>>>(y_out, edge, g_e, be_e);
}
